// GAC_Net_42314017800359
// MI455X (gfx1250) — matrix operand order repaired and checked by reading, not compiled, not run on hardware
//
#include <hip/hip_runtime.h>
#include <hip/hip_bf16.h>

// ---------------------------------------------------------------------------
// GAC-Net forward for MI455X (gfx1250, wave32, WMMA), round 2.
//
// Key change vs round 1: all GEMM operands are pre-converted to zero-padded
// f16 so WMMA A/B fragments are loaded *directly from global memory* as
// 16B/32B vectors in one contiguous k order shared by A and B (k = 16*(lane>>4) + s;
// not the ISA table order). The GEMM kernel has no LDS, no barriers, no bounds checks, and
// carries 4 accumulators per wave (A-fragment reused 4x).
//
// Input flattening assumption (setup_inputs() dict insertion order):
//   d_in[0]      points            (8,4096,9)  f32
//   d_in[1..4]   gac_mlp L1: W(64,6)   b(64)  gamma(64)  beta(64)
//   d_in[5..8]   gac_mlp L2: W(64,64)  b g beta
//   d_in[9..12]  gac_mlp L3: W(128,64) b g beta
//   d_in[13]     a                 (131,128)
//   d_in[14..17] fp_mlp L1: W(128,134) b g beta
//   d_in[18..21] fp_mlp L2: W(128,128) b g beta
//   d_in[22]     head_w            (13,128)
//   d_in[23]     head_b            (13)
// Output: logits (8,4096,13) f32.  Workspace ~480 MiB (bump allocator).
// ---------------------------------------------------------------------------

#define BATCH 8
#define NPTS  4096
#define NP    1024
#define NS    32
#define RAD2  0.0225f   /* 0.15^2 */
#define LALPHA 0.2f
#define BNEPS 1e-5f
#define BIGF  1e10f
#define NCLS  13

typedef __attribute__((ext_vector_type(16))) _Float16 v16h;
typedef __attribute__((ext_vector_type(8)))  _Float16 v8h;
typedef __attribute__((ext_vector_type(8)))  float    v8f;

// ---------------------------------------------------------------------------
// 1) Farthest point sampling: one block per batch; register-resident min
//    distances (16 pts/thread), LDS argmax tree. 1024 dependent steps.
// ---------------------------------------------------------------------------
__global__ __launch_bounds__(256) void k_fps(const float* __restrict__ pts,
                                             int* __restrict__ cidx) {
  const int b = blockIdx.x;
  const int t = threadIdx.x;
  __shared__ float sval[256];
  __shared__ int   sidx[256];
  __shared__ int   sfar;
  __shared__ float scx, scy, scz;
  float mind[16];
#pragma unroll
  for (int i = 0; i < 16; ++i) mind[i] = BIGF;
  int far = 0;
  for (int it = 0; it < NP; ++it) {
    if (t == 0) {
      cidx[b * NP + it] = far;
      const float* c = pts + (size_t)(b * NPTS + far) * 9;
      scx = c[0]; scy = c[1]; scz = c[2];
    }
    __syncthreads();
    const float cx = scx, cy = scy, cz = scz;
    float bestv = -1.0f; int besti = 0;
#pragma unroll
    for (int i = 0; i < 16; ++i) {
      const int p = t + i * 256;
      const float* q = pts + (size_t)(b * NPTS + p) * 9;
      const float dx = q[0] - cx, dy = q[1] - cy, dz = q[2] - cz;
      const float d = dx * dx + dy * dy + dz * dz;
      float m = mind[i];
      m = (d < m) ? d : m;
      mind[i] = m;
      if (m > bestv) { bestv = m; besti = p; }
    }
    sval[t] = bestv; sidx[t] = besti;
    __syncthreads();
    for (int off = 128; off > 0; off >>= 1) {
      if (t < off && sval[t + off] > sval[t]) {
        sval[t] = sval[t + off]; sidx[t] = sidx[t + off];
      }
      __syncthreads();
    }
    if (t == 0) sfar = sidx[0];
    __syncthreads();
    far = sfar;
  }
}

// ---------------------------------------------------------------------------
// 2) Gather centroids: cxyz f32 + cfeat directly as padded f16 (K 6 -> 32).
// ---------------------------------------------------------------------------
__global__ void k_gather_c(const float* __restrict__ pts, const int* __restrict__ cidx,
                           float* __restrict__ cxyz, _Float16* __restrict__ cfeatH) {
  const int c = blockIdx.x * blockDim.x + threadIdx.x;
  if (c >= BATCH * NP) return;
  const int b = c >> 10;
  const int j = cidx[c];
  const float* q = pts + (size_t)(b * NPTS + j) * 9;
  cxyz[c * 3 + 0] = q[0]; cxyz[c * 3 + 1] = q[1]; cxyz[c * 3 + 2] = q[2];
  _Float16* row = cfeatH + (size_t)c * 32;
#pragma unroll
  for (int k = 0; k < 6; ++k) row[k] = (_Float16)q[3 + k];
#pragma unroll
  for (int k = 6; k < 32; ++k) row[k] = (_Float16)0.f;
}

// ---------------------------------------------------------------------------
// 3) Ball query: thread/centroid, keep 32 smallest in-radius distances.
// ---------------------------------------------------------------------------
__global__ void k_query_ball(const float* __restrict__ pts, const float* __restrict__ cxyz,
                             int* __restrict__ gidx) {
  const int c = blockIdx.x * blockDim.x + threadIdx.x;
  if (c >= BATCH * NP) return;
  const int b = c >> 10;
  const float cx = cxyz[c * 3], cy = cxyz[c * 3 + 1], cz = cxyz[c * 3 + 2];
  float bd[NS]; int bi[NS]; int cnt = 0;
  for (int j = 0; j < NPTS; ++j) {
    const float* q = pts + (size_t)(b * NPTS + j) * 9;
    const float dx = q[0] - cx, dy = q[1] - cy, dz = q[2] - cz;
    const float d = dx * dx + dy * dy + dz * dz;
    if (d > RAD2) continue;
    if (cnt < NS) {
      int k = cnt++;
      while (k > 0 && bd[k - 1] > d) { bd[k] = bd[k - 1]; bi[k] = bi[k - 1]; --k; }
      bd[k] = d; bi[k] = j;
    } else if (d < bd[NS - 1]) {
      int k = NS - 1;
      while (k > 0 && bd[k - 1] > d) { bd[k] = bd[k - 1]; bi[k] = bi[k - 1]; --k; }
      bd[k] = d; bi[k] = j;
    }
  }
  const int f = (cnt > 0) ? bi[0] : 0;
  for (int k = cnt; k < NS; ++k) bi[k] = f;
  for (int k = 0; k < NS; ++k) gidx[c * NS + k] = bi[k];
}

// ---------------------------------------------------------------------------
// 4) Gather groups: gxyz f32 + gfeat as padded f16 (K 6 -> 32).
// ---------------------------------------------------------------------------
__global__ void k_gather_g(const float* __restrict__ pts, const int* __restrict__ gidx,
                           float* __restrict__ gxyz, _Float16* __restrict__ gfeatH) {
  const int e = blockIdx.x * blockDim.x + threadIdx.x;
  if (e >= BATCH * NP * NS) return;
  const int c = e / NS;
  const int b = c >> 10;
  const int j = gidx[e];
  const float* q = pts + (size_t)(b * NPTS + j) * 9;
  gxyz[(size_t)e * 3 + 0] = q[0];
  gxyz[(size_t)e * 3 + 1] = q[1];
  gxyz[(size_t)e * 3 + 2] = q[2];
  _Float16* row = gfeatH + (size_t)e * 32;
#pragma unroll
  for (int k = 0; k < 6; ++k) row[k] = (_Float16)q[3 + k];
#pragma unroll
  for (int k = 6; k < 32; ++k) row[k] = (_Float16)0.f;
}

__global__ void k_zero(float* __restrict__ p, int n) {
  const int i = blockIdx.x * blockDim.x + threadIdx.x;
  if (i < n) p[i] = 0.f;
}

// ---------------------------------------------------------------------------
// 5) Generic f32 -> padded f16 convert (used for weights): WH is (rowsP,Kp),
//    zero-filled outside (rows,K).
// ---------------------------------------------------------------------------
__global__ void k_cvt_pad(const float* __restrict__ X, _Float16* __restrict__ XH,
                          int rows, int K, int Kp, int rowsP) {
  const size_t i = (size_t)blockIdx.x * blockDim.x + threadIdx.x;
  if (i >= (size_t)rowsP * Kp) return;
  const int r = (int)(i / Kp), c = (int)(i % Kp);
  const float v = (r < rows && c < K) ? X[(size_t)r * K + c] : 0.f;
  XH[i] = (_Float16)v;
}

// a(131,128) -> aHT(128,160) f16, transposed so B-fragments are contiguous.
__global__ void k_cvt_aT(const float* __restrict__ a, _Float16* __restrict__ aHT) {
  const int i = blockIdx.x * blockDim.x + threadIdx.x;
  if (i >= 128 * 160) return;
  const int n = i / 160, k = i % 160;
  aHT[i] = (_Float16)((k < 131) ? a[(size_t)k * 128 + n] : 0.f);
}

// ---------------------------------------------------------------------------
// 6) WMMA GEMM, no LDS: Y[M,Cout] = Xh[M,Kp] @ Wh(CoutP,Kp)^T + bias.
//    Block = 256 thr (8 waves) covers 128 rows x 64 cols; each wave holds one
//    A-fragment and 4 accumulators. Fragments are direct vector loads:
//      A: lane l -> row (wv*16 + l&15), halves kb+(l>>4)*16 +{0..15}
//      B: lane l -> W row nBase+16*nt+(l&15), halves kb+(l>>4)*16 +{0..15}
//    Fused BN statistics (sum/sumsq) via shfl_xor(16) + atomics.
// ---------------------------------------------------------------------------
__global__ __launch_bounds__(256) void k_gemm_wmma(
    const _Float16* __restrict__ Xh, const _Float16* __restrict__ Wh,
    const float* __restrict__ bias, float* __restrict__ Y,
    int M, int Kp, int Cout, float* __restrict__ stats, int do_stats) {
  const int t = threadIdx.x;
  const int wv = t >> 5, lane = t & 31;
  const int mr = lane & 15;
  const int ah = (lane >> 4) << 4;     // A K offset within tile: 0 or 16
  const int bh = (lane >> 4) << 4;     // B K offset within tile: 0 or 16
  const int rowBlk = blockIdx.x * 128;
  const int nBase = blockIdx.y * 64;
  const _Float16* aPtr = Xh + (size_t)(rowBlk + (wv << 4) + mr) * Kp + ah;
  const _Float16* bPtr = Wh + (size_t)(nBase + mr) * Kp + bh;
  const size_t bStride = (size_t)16 * Kp;
  v8f acc0 = {}, acc1 = {}, acc2 = {}, acc3 = {};
  for (int kb = 0; kb < Kp; kb += 32) {
    __builtin_prefetch(aPtr + kb + 64, 0, 2);  // speculative -> safe past end
    v16h a;
    const v8h alo = *(const v8h*)(aPtr + kb);
    const v8h ahi = *(const v8h*)(aPtr + kb + 8);
#pragma unroll
    for (int j = 0; j < 8; ++j) { a[j] = alo[j]; a[8 + j] = ahi[j]; }
    const v16h b0 = *(const v16h*)(bPtr + kb);
    const v16h b1 = *(const v16h*)(bPtr + bStride + kb);
    const v16h b2 = *(const v16h*)(bPtr + 2 * bStride + kb);
    const v16h b3 = *(const v16h*)(bPtr + 3 * bStride + kb);
    acc0 = __builtin_amdgcn_wmma_f32_16x16x32_f16(false, a, false, b0, (short)0, acc0, false, false);
    acc1 = __builtin_amdgcn_wmma_f32_16x16x32_f16(false, a, false, b1, (short)0, acc1, false, false);
    acc2 = __builtin_amdgcn_wmma_f32_16x16x32_f16(false, a, false, b2, (short)0, acc2, false, false);
    acc3 = __builtin_amdgcn_wmma_f32_16x16x32_f16(false, a, false, b3, (short)0, acc3, false, false);
  }
  const int mB = rowBlk + (wv << 4) + ((lane >> 4) << 3);
  v8f accs[4] = {acc0, acc1, acc2, acc3};
#pragma unroll
  for (int nt = 0; nt < 4; ++nt) {
    const int col = nBase + nt * 16 + mr;
    const float bsv = (col < Cout) ? bias[col] : 0.f;
    float lsum = 0.f, lsq = 0.f;
#pragma unroll
    for (int j = 0; j < 8; ++j) {
      const float v = accs[nt][j] + bsv;
      if (col < Cout) Y[(size_t)(mB + j) * Cout + col] = v;
      lsum += v; lsq += v * v;
    }
    if (do_stats) {
      lsum += __shfl_xor(lsum, 16, 32);  // join M=0..7 / M=8..15 lane halves
      lsq  += __shfl_xor(lsq, 16, 32);
      if (lane < 16 && col < Cout) {
        atomicAdd(&stats[col], lsum);
        atomicAdd(&stats[256 + col], lsq);
      }
    }
  }
}

// ---------------------------------------------------------------------------
// 7) BN + ReLU apply; emits f32 in place and the next layer's f16 operand.
// ---------------------------------------------------------------------------
__global__ void k_bn_relu_cvt(float* __restrict__ Y, _Float16* __restrict__ Yh,
                              const float* __restrict__ stats,
                              const float* __restrict__ gamma, const float* __restrict__ beta,
                              int M, int Cout) {
  const size_t i = (size_t)blockIdx.x * blockDim.x + threadIdx.x;
  if (i >= (size_t)M * Cout) return;
  const int c = (int)(i % Cout);
  const float mean = stats[c] / (float)M;
  const float var = stats[256 + c] / (float)M - mean * mean;
  float v = (Y[i] - mean) * rsqrtf(var + BNEPS) * gamma[c] + beta[c];
  v = v > 0.f ? v : 0.f;
  Y[i] = v;
  Yh[i] = (_Float16)v;   // Cout is 64/128 here: already a multiple of 32
}

// ---------------------------------------------------------------------------
// 8) Fused GAC attention per centroid: delta(32x160 f16) in LDS, WMMA against
//    pre-transposed aHT(128x160 f16) with direct 32B B-fragment loads,
//    leaky-relu -> LDS, softmax over the 32 samples, weighted gf sum -> gac.
// ---------------------------------------------------------------------------
__global__ __launch_bounds__(256) void k_gac_att(
    const float* __restrict__ gxyz, const float* __restrict__ cxyz,
    const float* __restrict__ gf, const float* __restrict__ cf,
    const _Float16* __restrict__ aHT, float* __restrict__ gac) {
  __shared__ _Float16 Ds[32][168];   // delta, K padded 131 -> 160
  __shared__ float Att[32][128];
  const int c = blockIdx.x;
  const int t = threadIdx.x;
  const int wv = t >> 5, lane = t & 31;
  for (int e = t; e < 32 * 160; e += 256) {
    const int s = e / 160, k = e % 160;
    float v = 0.f;
    if (k < 3)
      v = gxyz[(size_t)(c * NS + s) * 3 + k] - cxyz[(size_t)c * 3 + k];
    else if (k < 131)
      v = gf[(size_t)(c * NS + s) * 128 + (k - 3)] - cf[(size_t)c * 128 + (k - 3)];
    Ds[s][k] = (_Float16)v;
  }
  __syncthreads();
  v8f acc0 = {}, acc1 = {};
  const int mr = lane & 15;
  const int ah = (lane >> 4) << 4;
  const int bh = (lane >> 4) << 4;
  const int n = (wv << 4) + mr;
  const _Float16* bPtr = aHT + (size_t)n * 160 + bh;
#pragma unroll
  for (int ks = 0; ks < 5; ++ks) {
    const int kb = ks << 5;
    v16h a0, a1;
#pragma unroll
    for (int j = 0; j < 8; ++j) {
      a0[j]     = Ds[mr][kb + ah + j];
      a0[8 + j] = Ds[mr][kb + 8 + ah + j];
      a1[j]     = Ds[16 + mr][kb + ah + j];
      a1[8 + j] = Ds[16 + mr][kb + 8 + ah + j];
    }
    const v16h bfr = *(const v16h*)(bPtr + kb);
    acc0 = __builtin_amdgcn_wmma_f32_16x16x32_f16(false, a0, false, bfr, (short)0, acc0, false, false);
    acc1 = __builtin_amdgcn_wmma_f32_16x16x32_f16(false, a1, false, bfr, (short)0, acc1, false, false);
  }
  const int sB = (lane >> 4) << 3;
#pragma unroll
  for (int j = 0; j < 8; ++j) {
    const float v0 = acc0[j]; Att[sB + j][n]      = (v0 > 0.f) ? v0 : LALPHA * v0;
    const float v1 = acc1[j]; Att[16 + sB + j][n] = (v1 > 0.f) ? v1 : LALPHA * v1;
  }
  __syncthreads();
  if (t < 128) {
    const int ch = t;
    float mx = -BIGF;
    for (int s = 0; s < NS; ++s) mx = fmaxf(mx, Att[s][ch]);
    float sum = 0.f, out = 0.f;
    for (int s = 0; s < NS; ++s) sum += __expf(Att[s][ch] - mx);
    for (int s = 0; s < NS; ++s)
      out += __expf(Att[s][ch] - mx) * gf[(size_t)(c * NS + s) * 128 + ch];
    gac[(size_t)c * 128 + ch] = out / sum;
  }
}

// ---------------------------------------------------------------------------
// 9) 3-NN interpolation + concat, writing fp_mlp's padded f16 input directly
//    (134 -> 160). Faithfully replicates dk = where(dk < BIG, 1e-10, dk).
// ---------------------------------------------------------------------------
__global__ void k_interp(const float* __restrict__ pts, const float* __restrict__ cxyz,
                         const float* __restrict__ gac, _Float16* __restrict__ nfH) {
  const int p = blockIdx.x * blockDim.x + threadIdx.x;
  if (p >= BATCH * NPTS) return;
  const int b = p >> 12;
  const float* q = pts + (size_t)p * 9;
  const float x = q[0], y = q[1], z = q[2];
  float d0 = BIGF, d1 = BIGF, d2 = BIGF; int i0 = 0, i1 = 0, i2 = 0;
  for (int cc = 0; cc < NP; ++cc) {
    const float* ctr = cxyz + (size_t)(b * NP + cc) * 3;
    const float dx = ctr[0] - x, dy = ctr[1] - y, dz = ctr[2] - z;
    const float d = dx * dx + dy * dy + dz * dz;
    if (d < d0)      { d2 = d1; i2 = i1; d1 = d0; i1 = i0; d0 = d; i0 = cc; }
    else if (d < d1) { d2 = d1; i2 = i1; d1 = d;  i1 = cc; }
    else if (d < d2) { d2 = d;  i2 = cc; }
  }
  const float w0 = 1.f / ((d0 < BIGF) ? 1e-10f : d0);
  const float w1 = 1.f / ((d1 < BIGF) ? 1e-10f : d1);
  const float w2 = 1.f / ((d2 < BIGF) ? 1e-10f : d2);
  const float ws = w0 + w1 + w2;
  _Float16* row = nfH + (size_t)p * 160;
#pragma unroll
  for (int k = 0; k < 6; ++k) row[k] = (_Float16)q[3 + k];
  const float* g0 = gac + (size_t)(b * NP + i0) * 128;
  const float* g1 = gac + (size_t)(b * NP + i1) * 128;
  const float* g2 = gac + (size_t)(b * NP + i2) * 128;
  for (int ch = 0; ch < 128; ++ch)
    row[6 + ch] = (_Float16)((w0 * g0[ch] + w1 * g1[ch] + w2 * g2[ch]) / ws);
#pragma unroll
  for (int k = 134; k < 160; ++k) row[k] = (_Float16)0.f;
}

// ---------------------------------------------------------------------------
// Host-side orchestration (graph-capture safe: only stream launches).
// ---------------------------------------------------------------------------
extern "C" void kernel_launch(void* const* d_in, const int* in_sizes, int n_in,
                              void* d_out, int out_size, void* d_ws, size_t ws_size,
                              hipStream_t stream) {
  (void)in_sizes; (void)n_in; (void)out_size; (void)ws_size;
  const float* points = (const float*)d_in[0];
  const float* gW[3]  = {(const float*)d_in[1], (const float*)d_in[5], (const float*)d_in[9]};
  const float* gB[3]  = {(const float*)d_in[2], (const float*)d_in[6], (const float*)d_in[10]};
  const float* gG[3]  = {(const float*)d_in[3], (const float*)d_in[7], (const float*)d_in[11]};
  const float* gBe[3] = {(const float*)d_in[4], (const float*)d_in[8], (const float*)d_in[12]};
  const float* aMat   = (const float*)d_in[13];
  const float* fW[2]  = {(const float*)d_in[14], (const float*)d_in[18]};
  const float* fB[2]  = {(const float*)d_in[15], (const float*)d_in[19]};
  const float* fG[2]  = {(const float*)d_in[16], (const float*)d_in[20]};
  const float* fBe[2] = {(const float*)d_in[17], (const float*)d_in[21]};
  const float* headW  = (const float*)d_in[22];
  const float* headB  = (const float*)d_in[23];
  float* out = (float*)d_out;

  const int MC = BATCH * NP;        // 8192 centroid rows
  const int MG = BATCH * NP * NS;   // 262144 grouped rows
  const int MN = BATCH * NPTS;      // 32768 point rows

  char* wp = (char*)d_ws;
  auto alloc = [&](size_t bytes) -> void* {
    void* p = (void*)wp;
    wp += (bytes + 255) & ~(size_t)255;
    return p;
  };
  // indices / geometry
  int*   cidx  = (int*)alloc((size_t)MC * 4);
  int*   gidx  = (int*)alloc((size_t)MG * 4);
  float* cxyz  = (float*)alloc((size_t)MC * 3 * 4);
  float* gxyz  = (float*)alloc((size_t)MG * 3 * 4);
  // padded f16 GEMM operands
  _Float16* cfeatH = (_Float16*)alloc((size_t)MC * 32 * 2);
  _Float16* gfeatH = (_Float16*)alloc((size_t)MG * 32 * 2);
  _Float16* cfAH   = (_Float16*)alloc((size_t)MC * 64 * 2);
  _Float16* cfBH   = (_Float16*)alloc((size_t)MC * 64 * 2);
  _Float16* cfOutH = (_Float16*)alloc((size_t)MC * 128 * 2);
  _Float16* gfAH   = (_Float16*)alloc((size_t)MG * 64 * 2);
  _Float16* gfBH   = (_Float16*)alloc((size_t)MG * 64 * 2);
  _Float16* gfOutH = (_Float16*)alloc((size_t)MG * 128 * 2);
  _Float16* nfH    = (_Float16*)alloc((size_t)MN * 160 * 2);
  _Float16* nfAH   = (_Float16*)alloc((size_t)MN * 128 * 2);
  _Float16* nfBH   = (_Float16*)alloc((size_t)MN * 128 * 2);
  // f32 layer outputs (pre/post BN)
  float* cfA   = (float*)alloc((size_t)MC * 64 * 4);
  float* cfBuf = (float*)alloc((size_t)MC * 64 * 4);
  float* cfOut = (float*)alloc((size_t)MC * 128 * 4);
  float* gfA   = (float*)alloc((size_t)MG * 64 * 4);
  float* gfB   = (float*)alloc((size_t)MG * 64 * 4);
  float* gfOut = (float*)alloc((size_t)MG * 128 * 4);
  float* gacF  = (float*)alloc((size_t)MC * 128 * 4);
  float* nfA   = (float*)alloc((size_t)MN * 128 * 4);
  float* nfB   = (float*)alloc((size_t)MN * 128 * 4);
  float* stats = (float*)alloc(512 * 4);
  // padded f16 weights: (CoutP, Kp)
  _Float16* g1H   = (_Float16*)alloc((size_t)64 * 32 * 2);
  _Float16* g2H   = (_Float16*)alloc((size_t)64 * 64 * 2);
  _Float16* g3H   = (_Float16*)alloc((size_t)128 * 64 * 2);
  _Float16* f1H   = (_Float16*)alloc((size_t)128 * 160 * 2);
  _Float16* f2H   = (_Float16*)alloc((size_t)128 * 128 * 2);
  _Float16* hH    = (_Float16*)alloc((size_t)64 * 128 * 2);
  _Float16* aHT   = (_Float16*)alloc((size_t)128 * 160 * 2);

  // --- one-time weight conversions (padded f16) ---
  auto cvtW = [&](const float* W, _Float16* WH, int Cout, int K, int Kp, int CoutP) {
    const size_t tot = (size_t)CoutP * Kp;
    k_cvt_pad<<<(unsigned)((tot + 255) / 256), 256, 0, stream>>>(W, WH, Cout, K, Kp, CoutP);
  };
  cvtW(gW[0], g1H, 64, 6, 32, 64);
  cvtW(gW[1], g2H, 64, 64, 64, 64);
  cvtW(gW[2], g3H, 128, 64, 64, 128);
  cvtW(fW[0], f1H, 128, 134, 160, 128);
  cvtW(fW[1], f2H, 128, 128, 128, 128);
  cvtW(headW, hH, 13, 128, 128, 64);
  k_cvt_aT<<<(128 * 160) / 256, 256, 0, stream>>>(aMat, aHT);

  // --- sampling / grouping ---
  k_fps<<<BATCH, 256, 0, stream>>>(points, cidx);
  k_gather_c<<<MC / 256, 256, 0, stream>>>(points, cidx, cxyz, cfeatH);
  k_query_ball<<<MC / 256, 256, 0, stream>>>(points, cxyz, gidx);
  k_gather_g<<<MG / 256, 256, 0, stream>>>(points, gidx, gxyz, gfeatH);

  // --- GEMM + BN + ReLU layer helper ---
  auto layer = [&](const _Float16* XH, const _Float16* WH, const float* b,
                   const float* gamma, const float* beta,
                   float* Y, _Float16* YH, int M, int Kp, int Cout) {
    k_zero<<<2, 256, 0, stream>>>(stats, 512);
    dim3 grid(M / 128, (Cout + 63) / 64);
    k_gemm_wmma<<<grid, 256, 0, stream>>>(XH, WH, b, Y, M, Kp, Cout, stats, 1);
    const size_t tot = (size_t)M * Cout;
    k_bn_relu_cvt<<<(unsigned)((tot + 255) / 256), 256, 0, stream>>>(Y, YH, stats, gamma, beta, M, Cout);
  };

  // --- gac_mlp on centroid features (M=8192) ---
  layer(cfeatH, g1H, gB[0], gG[0], gBe[0], cfA,   cfAH,   MC, 32, 64);
  layer(cfAH,   g2H, gB[1], gG[1], gBe[1], cfBuf, cfBH,   MC, 64, 64);
  layer(cfBH,   g3H, gB[2], gG[2], gBe[2], cfOut, cfOutH, MC, 64, 128);

  // --- gac_mlp on grouped features (M=262144) ---
  layer(gfeatH, g1H, gB[0], gG[0], gBe[0], gfA,   gfAH,   MG, 32, 64);
  layer(gfAH,   g2H, gB[1], gG[1], gBe[1], gfB,   gfBH,   MG, 64, 64);
  layer(gfBH,   g3H, gB[2], gG[2], gBe[2], gfOut, gfOutH, MG, 64, 128);

  // --- fused attention + softmax + aggregation ---
  k_gac_att<<<MC, 256, 0, stream>>>(gxyz, cxyz, gfOut, cfOut, aHT, gacF);

  // --- 3-NN interpolation + concat -> padded f16 fp_mlp input ---
  k_interp<<<MN / 256, 256, 0, stream>>>(points, cxyz, gacF, nfH);

  // --- fp_mlp (M=32768) ---
  layer(nfH,  f1H, fB[0], fG[0], fBe[0], nfA, nfAH, MN, 160, 128);
  layer(nfAH, f2H, fB[1], fG[1], fBe[1], nfB, nfBH, MN, 128, 128);

  // --- head: logits = nfB @ head_w^T + head_b (no BN/ReLU, stats off) ---
  {
    dim3 grid(MN / 128, 1);
    k_gemm_wmma<<<grid, 256, 0, stream>>>(nfBH, hH, headB, out, MN, 128, NCLS, stats, 0);
  }
}
